// SimpleRGCN_15547781611628
// MI455X (gfx1250) — matrix operand order repaired and checked by reading, compiled, not run on hardware
//
#include <hip/hip_runtime.h>
#include <hip/hip_bf16.h>

#define N_NODES 100000
#define N_EDGES 3200000
#define N_REL   8
#define EMB     128

typedef __attribute__((ext_vector_type(16))) __bf16 v16bf;
typedef __attribute__((ext_vector_type(8)))  __bf16 v8bf;
typedef __attribute__((ext_vector_type(4)))  __bf16 v4bf;
typedef __attribute__((ext_vector_type(8)))  float  v8f;

// ---------------------------------------------------------------------------
// Zero workspace (deg[800K] + agg[102.4M] floats)
// ---------------------------------------------------------------------------
__global__ __launch_bounds__(256) void rgcn_zero(float4* __restrict__ p, int n4) {
    int i = blockIdx.x * 256 + threadIdx.x;
    if (i < n4) p[i] = make_float4(0.f, 0.f, 0.f, 0.f);
}

// ---------------------------------------------------------------------------
// Degree count: deg[rel*N + src] += 1
// ---------------------------------------------------------------------------
__global__ __launch_bounds__(256) void rgcn_degree(const int* __restrict__ src,
                                                   const int* __restrict__ rel,
                                                   float* __restrict__ deg) {
    int e = blockIdx.x * 256 + threadIdx.x;
    if (e < N_EDGES) {
        int row = src[e] + N_NODES * rel[e];
        unsafeAtomicAdd(&deg[row], 1.0f);
    }
}

// ---------------------------------------------------------------------------
// Scatter: agg[row][:] += node_emb[dst][:]   (one wave32 per edge, float4/lane)
// ---------------------------------------------------------------------------
__global__ __launch_bounds__(256) void rgcn_scatter(const float* __restrict__ node_emb,
                                                    const int* __restrict__ src,
                                                    const int* __restrict__ rel,
                                                    const int* __restrict__ dst,
                                                    float* __restrict__ agg) {
    int tid  = threadIdx.x;
    int wave = tid >> 5;
    int lane = tid & 31;
    int e = blockIdx.x * 8 + wave;
    if (e >= N_EDGES) return;
    int row = src[e] + N_NODES * rel[e];
    int d   = dst[e];
    float4 v = *reinterpret_cast<const float4*>(node_emb + (size_t)d * EMB + lane * 4);
    float* a = agg + (size_t)row * EMB + lane * 4;
    unsafeAtomicAdd(a + 0, v.x);
    unsafeAtomicAdd(a + 1, v.y);
    unsafeAtomicAdd(a + 2, v.z);
    unsafeAtomicAdd(a + 3, v.w);
}

// ---------------------------------------------------------------------------
// GEMM: out[n,i] = relu( sum_{r,j} W[r,i,j] * (agg[r,n,j] * invdeg[r,n]) )
// M = N_NODES, K = N_REL*EMB = 1024, N = EMB = 128.
// bf16 hi/lo split (3 WMMAs) for fp32-class accuracy on matrix cores.
// ---------------------------------------------------------------------------
#define LDSP 40   // row pitch in bf16 elements (80B = 20 DWORDs, conflict-free)

static __device__ inline v16bf join16(v8bf a, v8bf b) {
    return __builtin_shufflevector(a, b, 0,1,2,3,4,5,6,7,8,9,10,11,12,13,14,15);
}
static __device__ inline v8bf ld8(const __bf16* p) {
    return *reinterpret_cast<const v8bf*>(p);
}

__global__ __launch_bounds__(256) void rgcn_gemm(const float* __restrict__ agg,
                                                 const float* __restrict__ deg,
                                                 const float* __restrict__ W,
                                                 float* __restrict__ out) {
    __shared__ alignas(16) __bf16 Ah[128 * LDSP];
    __shared__ alignas(16) __bf16 Al[128 * LDSP];
    __shared__ alignas(16) __bf16 Bh[128 * LDSP];
    __shared__ alignas(16) __bf16 Bl[128 * LDSP];

    const int tid  = threadIdx.x;
    const int wave = tid >> 5;
    const int lane = tid & 31;
    const int m0   = blockIdx.x * 128;

    v8f acc[8] = {};

    for (int kc = 0; kc < 32; ++kc) {          // 32 K-chunks of 32
        const int r  = kc >> 2;                // relation
        const int j0 = (kc & 3) * 32;          // offset within EMB

        // ---- stage A (agg, scaled by inv-degree) and B (weights) into LDS ----
        #pragma unroll
        for (int it = 0; it < 4; ++it) {
            int idx  = tid + it * 256;         // 0..1023
            int row  = idx >> 3;               // 0..127
            int c0   = (idx & 7) * 4;          // 0,4,...,28

            // A: row = node within tile
            float4 va = make_float4(0.f, 0.f, 0.f, 0.f);
            int node = m0 + row;
            if (node < N_NODES) {
                float dg = deg[r * N_NODES + node];
                float s  = dg > 0.f ? 1.0f / dg : 0.f;
                va = *reinterpret_cast<const float4*>(
                        agg + ((size_t)r * N_NODES + node) * EMB + j0 + c0);
                va.x *= s; va.y *= s; va.z *= s; va.w *= s;
            }
            __bf16 h0 = (__bf16)va.x, h1 = (__bf16)va.y, h2 = (__bf16)va.z, h3 = (__bf16)va.w;
            __bf16 l0 = (__bf16)(va.x - (float)h0), l1 = (__bf16)(va.y - (float)h1);
            __bf16 l2 = (__bf16)(va.z - (float)h2), l3 = (__bf16)(va.w - (float)h3);
            *reinterpret_cast<v4bf*>(&Ah[row * LDSP + c0]) = (v4bf){h0, h1, h2, h3};
            *reinterpret_cast<v4bf*>(&Al[row * LDSP + c0]) = (v4bf){l0, l1, l2, l3};

            // B: row = output feature i; B[k=j][i] = W[r][i][j]
            float4 vb = *reinterpret_cast<const float4*>(
                    W + (size_t)r * (EMB * EMB) + (size_t)row * EMB + j0 + c0);
            __bf16 g0 = (__bf16)vb.x, g1 = (__bf16)vb.y, g2 = (__bf16)vb.z, g3 = (__bf16)vb.w;
            __bf16 m0b = (__bf16)(vb.x - (float)g0), m1b = (__bf16)(vb.y - (float)g1);
            __bf16 m2b = (__bf16)(vb.z - (float)g2), m3b = (__bf16)(vb.w - (float)g3);
            *reinterpret_cast<v4bf*>(&Bh[row * LDSP + c0]) = (v4bf){g0, g1, g2, g3};
            *reinterpret_cast<v4bf*>(&Bl[row * LDSP + c0]) = (v4bf){m0b, m1b, m2b, m3b};
        }
        __syncthreads();

        // ---- A fragments (16-bit A 16x32 layout) ----
        const int mrow = wave * 16 + (lane & 15);
        const int kofA = (lane >> 4) * 8;      // lanes 0-15: K0-7/16-23; 16-31: K8-15/24-31
        v16bf aH = join16(ld8(&Ah[mrow * LDSP + kofA]), ld8(&Ah[mrow * LDSP + kofA + 16]));
        v16bf aL = join16(ld8(&Al[mrow * LDSP + kofA]), ld8(&Al[mrow * LDSP + kofA + 16]));

        // ---- B fragments + WMMA over 8 N-tiles ----
        const int kb = (lane >> 4) * 8;        // lanes 0-15: K0-7/16-23; 16-31: K8-15/24-31
        #pragma unroll
        for (int nt = 0; nt < 8; ++nt) {
            int nrow = nt * 16 + (lane & 15);
            v16bf bH = join16(ld8(&Bh[nrow * LDSP + kb]), ld8(&Bh[nrow * LDSP + kb + 16]));
            v16bf bL = join16(ld8(&Bl[nrow * LDSP + kb]), ld8(&Bl[nrow * LDSP + kb + 16]));
            acc[nt] = __builtin_amdgcn_wmma_f32_16x16x32_bf16(
                false, aH, false, bH, (short)0, acc[nt], false, false);
            acc[nt] = __builtin_amdgcn_wmma_f32_16x16x32_bf16(
                false, aH, false, bL, (short)0, acc[nt], false, false);
            acc[nt] = __builtin_amdgcn_wmma_f32_16x16x32_bf16(
                false, aL, false, bH, (short)0, acc[nt], false, false);
        }
        __syncthreads();
    }

    // ---- epilogue: ReLU + store (C 16x16 f32 layout) ----
    const int mbase = wave * 16 + (lane >> 4) * 8;
    const int nlo   = lane & 15;
    #pragma unroll
    for (int nt = 0; nt < 8; ++nt) {
        #pragma unroll
        for (int v = 0; v < 8; ++v) {
            int node = m0 + mbase + v;
            if (node < N_NODES)
                out[(size_t)node * EMB + nt * 16 + nlo] = fmaxf(acc[nt][v], 0.f);
        }
    }
}

// ---------------------------------------------------------------------------
extern "C" void kernel_launch(void* const* d_in, const int* in_sizes, int n_in,
                              void* d_out, int out_size, void* d_ws, size_t ws_size,
                              hipStream_t stream) {
    const float* node_emb = (const float*)d_in[0];   // [100000,128]
    const float* weights  = (const float*)d_in[1];   // [8,128,128]
    const int*   src      = (const int*)d_in[2];     // [3.2M]
    const int*   rel      = (const int*)d_in[3];     // [3.2M]
    const int*   dst      = (const int*)d_in[4];     // [3.2M]
    float*       out      = (float*)d_out;           // [100000,128]

    float* deg = (float*)d_ws;                       // 800,000 floats
    float* agg = deg + (size_t)N_REL * N_NODES;      // 102,400,000 floats

    // zero deg + agg (103,200,000 floats = 25,800,000 float4)
    const int n4 = (N_REL * N_NODES + N_REL * N_NODES * EMB) / 4;
    rgcn_zero<<<(n4 + 255) / 256, 256, 0, stream>>>((float4*)d_ws, n4);

    // degree counts
    rgcn_degree<<<(N_EDGES + 255) / 256, 256, 0, stream>>>(src, rel, deg);

    // scatter unnormalized messages (one wave per edge)
    rgcn_scatter<<<N_EDGES / 8, 256, 0, stream>>>(node_emb, src, rel, dst, agg);

    // fused normalize + relation-transform + sum + relu via WMMA
    rgcn_gemm<<<(N_NODES + 127) / 128, 256, 0, stream>>>(agg, deg, weights, out);
}
